// TCE_19121194402030
// MI455X (gfx1250) — matrix operand order repaired and checked by reading, not compiled, not run on hardware
//
#include <hip/hip_runtime.h>
#include <hip/hip_bf16.h>
#include <math.h>

typedef __attribute__((ext_vector_type(16))) __bf16 v16bf;
typedef __attribute__((ext_vector_type(8)))  float  v8f;

#define B_    64
#define C_    30
#define L_    200
#define CL_   6000
#define KCONV 7
#define EPSBN 1e-5f

__device__ __forceinline__ float sigmoidf_(float x) {
    return 1.0f / (1.0f + __expf(-x));
}

// ---------------------------------------------------------------------------
// Causal conv1d: y[b,o,l] = bias[o] + sum_{i,k} x[b,i,l-6+k] * w[o,i,k]
// One block per batch element. X tile + all weights staged in LDS (~49 KB).
// ---------------------------------------------------------------------------
__global__ __launch_bounds__(256) void causal_conv_kernel(
    const float* __restrict__ X, const float* __restrict__ W,
    const float* __restrict__ bias, float* __restrict__ Y)
{
    __shared__ float xs[C_ * L_];
    __shared__ float ws[C_ * C_ * KCONV];
    const int b = blockIdx.x, tid = threadIdx.x;
    for (int i = tid; i < C_ * L_; i += 256) xs[i] = X[b * CL_ + i];
    for (int i = tid; i < C_ * C_ * KCONV; i += 256) ws[i] = W[i];
    __syncthreads();
    for (int idx = tid; idx < C_ * L_; idx += 256) {
        const int o = idx / L_, l = idx % L_;
        float acc = bias[o];
        #pragma unroll
        for (int k = 0; k < KCONV; ++k) {
            const int pos = l - (KCONV - 1) + k;
            if (pos >= 0) {
                for (int i = 0; i < C_; ++i)
                    acc += xs[i * L_ + pos] * ws[(o * C_ + i) * KCONV + k];
            }
        }
        Y[b * CL_ + idx] = acc;
    }
}

// ---------------------------------------------------------------------------
// Sigmoid attention (H heads, contiguous L-chunks), fused +X residual.
// y = sigmoid(((Q Kt)/sqrt(L)) V) + X.  One block per batch element.
// ---------------------------------------------------------------------------
__global__ __launch_bounds__(256) void attn_kernel(
    const float* __restrict__ Q, const float* __restrict__ Kk,
    const float* __restrict__ V, const float* __restrict__ X,
    float* __restrict__ Y, int H)
{
    __shared__ float qs[C_ * L_];
    __shared__ float ks[C_ * L_];
    __shared__ float ss[C_ * C_];
    const int b = blockIdx.x, tid = threadIdx.x;
    const float scale = rsqrtf((float)L_);   // scale uses full L (as in source)
    for (int i = tid; i < C_ * L_; i += 256) {
        qs[i] = Q[b * CL_ + i];
        ks[i] = Kk[b * CL_ + i];
    }
    __syncthreads();
    const int Lh = L_ / H;
    for (int h = 0; h < H; ++h) {
        const int base = h * Lh;
        for (int idx = tid; idx < C_ * C_; idx += 256) {
            const int c = idx / C_, d = idx % C_;
            float acc = 0.f;
            for (int t = 0; t < Lh; ++t)
                acc += qs[c * L_ + base + t] * ks[d * L_ + base + t];
            ss[idx] = acc * scale;
        }
        __syncthreads();
        for (int idx = tid; idx < C_ * Lh; idx += 256) {
            const int c = idx / Lh, l = base + idx % Lh;
            float acc = 0.f;
            for (int d = 0; d < C_; ++d)
                acc += ss[c * C_ + d] * V[b * CL_ + d * L_ + l];
            Y[b * CL_ + c * L_ + l] = sigmoidf_(acc) + X[b * CL_ + c * L_ + l];
        }
        __syncthreads();
    }
}

// ---------------------------------------------------------------------------
// BatchNorm1d (training stats, biased var) over (B,L) per channel.
// Optional second input added elementwise before normalization.
// Output written at out[b*out_bstride + out_coloff + c*L + l].
// One block per channel.
// ---------------------------------------------------------------------------
__global__ __launch_bounds__(256) void bn_kernel(
    const float* __restrict__ in, const float* __restrict__ in2,
    const float* __restrict__ g, const float* __restrict__ bt,
    float* __restrict__ out, int out_bstride, int out_coloff)
{
    __shared__ float red[256], red2[256];
    __shared__ float sm, sv;
    const int c = blockIdx.x, tid = threadIdx.x;
    float s1 = 0.f, s2 = 0.f;
    for (int i = tid; i < B_ * L_; i += 256) {
        const int b = i / L_, l = i % L_;
        const int off = b * CL_ + c * L_ + l;
        float x = in[off];
        if (in2) x += in2[off];
        s1 += x; s2 += x * x;
    }
    red[tid] = s1; red2[tid] = s2;
    __syncthreads();
    for (int s = 128; s > 0; s >>= 1) {
        if (tid < s) { red[tid] += red[tid + s]; red2[tid] += red2[tid + s]; }
        __syncthreads();
    }
    if (tid == 0) {
        const float m = red[0] / (float)(B_ * L_);
        const float v = red2[0] / (float)(B_ * L_) - m * m;
        sm = m; sv = rsqrtf(v + EPSBN);
    }
    __syncthreads();
    const float m = sm, inv = sv, gg = g[c], bb = bt[c];
    for (int i = tid; i < B_ * L_; i += 256) {
        const int b = i / L_, l = i % L_;
        const int off = b * CL_ + c * L_ + l;
        float x = in[off];
        if (in2) x += in2[off];
        out[(size_t)b * out_bstride + out_coloff + c * L_ + l] =
            gg * (x - m) * inv + bb;
    }
}

// ---------------------------------------------------------------------------
// GEMM helpers: tile staging with fast (vectorized, unguarded) and slow
// (scalar, guarded) paths.  KP=36 halves (72 B row stride) keeps packed bf16
// uint2 LDS stores 8-byte aligned and fragment dword reads conflict-free.
// ---------------------------------------------------------------------------
#define KP_ 36

template <int IT>
__device__ __forceinline__ void tile_load_regs(
    const float* __restrict__ g, int rbase, int k0, int R, int K,
    int tid, bool full, float4* regs)
{
    #pragma unroll
    for (int r = 0; r < IT; ++r) {
        const int idx = tid + r * 256;
        const int m = idx >> 3, kq = idx & 7;      // 8 float4 per 32-wide row
        const int row = rbase + m, col = k0 + kq * 4;
        if (full) {
            regs[r] = *(const float4*)&g[(size_t)row * K + col];
        } else {
            float4 v;
            v.x = (row < R && col + 0 < K) ? g[(size_t)row * K + col + 0] : 0.f;
            v.y = (row < R && col + 1 < K) ? g[(size_t)row * K + col + 1] : 0.f;
            v.z = (row < R && col + 2 < K) ? g[(size_t)row * K + col + 2] : 0.f;
            v.w = (row < R && col + 3 < K) ? g[(size_t)row * K + col + 3] : 0.f;
            regs[r] = v;
        }
    }
}

template <int IT, int ROWS>
__device__ __forceinline__ void tile_store_lds(
    const float4* regs, __bf16 (*S)[KP_], int tid)
{
    #pragma unroll
    for (int r = 0; r < IT; ++r) {
        const int idx = tid + r * 256;
        const int m = idx >> 3, kq = idx & 7;
        union { uint2 u2; __bf16 h[4]; } cv;
        cv.h[0] = (__bf16)regs[r].x; cv.h[1] = (__bf16)regs[r].y;
        cv.h[2] = (__bf16)regs[r].z; cv.h[3] = (__bf16)regs[r].w;
        *(uint2*)&S[m][kq * 4] = cv.u2;            // 8B-aligned ds_store_b64
    }
}

// ---------------------------------------------------------------------------
// D = A @ W^T + bias, A:[M,K] row-major, W:[N,K] row-major, both fp32 in HBM.
// fp32 -> bf16 while staging into LDS; compute uses v_wmma_f32_16x16x32_bf16
// with f32 accumulate.  Block = 256 threads = 8 waves.  BM covers the whole M
// so each weight row is streamed from HBM exactly once (activations ~3 MB are
// re-read from L2).  Register-pipelined LDS double buffering: next tile's
// global loads are issued before the current tile's WMMAs, so the loadcnt
// wait lands after the matrix work.
// ---------------------------------------------------------------------------
template <int BM, int WM, int WN, int TN>
__global__ __launch_bounds__(256) void gemm_wmma_kernel(
    const float* __restrict__ A, const float* __restrict__ W,
    const float* __restrict__ bias, float* __restrict__ D,
    int M, int N, int K)
{
    constexpr int BN  = WN * TN * 16;
    constexpr int AIT = (BM * 32) / (256 * 4);     // float4 per thread (A tile)
    constexpr int WIT = (BN * 32) / (256 * 4);     // float4 per thread (W tile)
    __shared__ __align__(16) __bf16 As[2][BM][KP_];
    __shared__ __align__(16) __bf16 Ws[2][BN][KP_];

    const int tid  = threadIdx.x;
    const int wave = tid >> 5, lane = tid & 31;
    const int wm = wave / WN, wn = wave % WN;
    const int lm = lane & 15, kh = lane >> 4;
    const int mbase = blockIdx.y * BM;
    const int nbase = blockIdx.x * BN;
    const bool rowsFullA = (mbase + BM <= M);
    const bool rowsFullW = (nbase + BN <= N);

    v8f acc[TN] = {};
    float4 aReg[AIT], wReg[WIT];

    const int nk = (K + 31) >> 5;

    // preload tile 0 into buffer 0
    {
        const bool kf = (32 <= K);
        tile_load_regs<AIT>(A, mbase, 0, M, K, tid, rowsFullA && kf, aReg);
        tile_load_regs<WIT>(W, nbase, 0, N, K, tid, rowsFullW && kf, wReg);
        tile_store_lds<AIT, BM>(aReg, As[0], tid);
        tile_store_lds<WIT, BN>(wReg, Ws[0], tid);
    }

    for (int it = 0; it < nk; ++it) {
        const int buf = it & 1;
        __syncthreads();                           // tile `it` visible in LDS

        const bool haveNext = (it + 1) < nk;
        if (haveNext) {
            const int k1 = (it + 1) << 5;
            const bool kf = (k1 + 32 <= K);
            tile_load_regs<AIT>(A, mbase, k1, M, K, tid, rowsFullA && kf, aReg);
            tile_load_regs<WIT>(W, nbase, k1, N, K, tid, rowsFullW && kf, wReg);
            if (it + 2 < nk) {                     // two tiles ahead: prefetch W
                const int n = nbase + (tid & (BN - 1));
                if (n < N) __builtin_prefetch(&W[(size_t)n * K + k1 + 32], 0, 1);
            }
        }

        // ---- compute tile `it` from LDS while next tile's loads are in flight
        v16bf af = {};
        #pragma unroll
        for (int j = 0; j < 8; ++j) {              // 16x32 bf16 A layout
            const int kb = (j < 4 ? 2 * j : 2 * j + 8) + 8 * kh;
            union { unsigned u; __bf16 h[2]; } cv;
            cv.u = *(const unsigned*)&As[buf][wm * 16 + lm][kb];
            af[2 * j] = cv.h[0]; af[2 * j + 1] = cv.h[1];
        }
        #pragma unroll
        for (int t = 0; t < TN; ++t) {
            v16bf bfr = {};
            const int nb = wn * TN * 16 + t * 16 + lm;
            #pragma unroll
            for (int j = 0; j < 8; ++j) {          // 32x16 bf16 B layout
                const int kb = (j < 4 ? 2 * j : 2 * j + 8) + 8 * kh;
                union { unsigned u; __bf16 h[2]; } cv;
                cv.u = *(const unsigned*)&Ws[buf][nb][kb];
                bfr[2 * j] = cv.h[0]; bfr[2 * j + 1] = cv.h[1];
            }
            acc[t] = __builtin_amdgcn_wmma_f32_16x16x32_bf16(
                false, af, false, bfr, (short)0, acc[t], false, false);
        }

        // ---- stash next tile into the other buffer (safe: it was consumed
        //      in iteration it-1, before this iteration's barrier)
        if (haveNext) {
            tile_store_lds<AIT, BM>(aReg, As[buf ^ 1], tid);
            tile_store_lds<WIT, BN>(wReg, Ws[buf ^ 1], tid);
        }
    }

    // epilogue: C/D layout -> lane col = lm, VGPR v -> row v + 8*kh
    #pragma unroll
    for (int t = 0; t < TN; ++t) {
        const int col = nbase + wn * TN * 16 + t * 16 + lm;
        if (col < N) {
            const float bv = bias ? bias[col] : 0.f;
            #pragma unroll
            for (int v = 0; v < 8; ++v) {
                const int row = mbase + wm * 16 + v + 8 * kh;
                if (row < M) D[(size_t)row * N + col] = acc[t][v] + bv;
            }
        }
    }
}

// ---------------------------------------------------------------------------
extern "C" void kernel_launch(void* const* d_in, const int* in_sizes, int n_in,
                              void* d_out, int out_size, void* d_ws, size_t ws_size,
                              hipStream_t stream)
{
    (void)in_sizes; (void)n_in; (void)out_size; (void)ws_size;

    const float* X = (const float*)d_in[0];
    const float* cwv[6] = {(const float*)d_in[1], (const float*)d_in[3],
                           (const float*)d_in[5], (const float*)d_in[7],
                           (const float*)d_in[9], (const float*)d_in[11]};
    const float* cbv[6] = {(const float*)d_in[2], (const float*)d_in[4],
                           (const float*)d_in[6], (const float*)d_in[8],
                           (const float*)d_in[10], (const float*)d_in[12]};
    const float* bn11g = (const float*)d_in[13]; const float* bn11b = (const float*)d_in[14];
    const float* bn12g = (const float*)d_in[15]; const float* bn12b = (const float*)d_in[16];
    const float* bn21g = (const float*)d_in[17]; const float* bn21b = (const float*)d_in[18];
    const float* bn22g = (const float*)d_in[19]; const float* bn22b = (const float*)d_in[20];
    const float* lin1w = (const float*)d_in[21]; const float* lin1b = (const float*)d_in[22];
    const float* lin2w = (const float*)d_in[23]; const float* lin2b = (const float*)d_in[24];
    const float* cw    = (const float*)d_in[25]; const float* cb    = (const float*)d_in[26];

    float* ws = (float*)d_ws;
    const size_t T1 = (size_t)B_ * CL_;        // 384000 floats
    float* qkv[6];
    for (int i = 0; i < 6; ++i) qkv[i] = ws + i * T1;
    float* y1   = ws + 6  * T1;                // attn1 + X
    float* y2   = ws + 7  * T1;                // attn2 + X
    float* Astk = ws + 8  * T1;                // [128,6000] bn11(y1) ++ bn21(y2)
    float* Tbuf = ws + 10 * T1;                // [128,6000] after lin1
    float* Obuf = ws + 12 * T1;                // [128,6000] after lin2
    float* cat  = ws + 14 * T1;                // [64,12000] out1 ++ out2
    float* outp = (float*)d_out;               // [64,6000]

    // 6 causal convs (Q1,K1,V1,Q2,K2,V2)
    for (int i = 0; i < 6; ++i)
        causal_conv_kernel<<<B_, 256, 0, stream>>>(X, cwv[i], cbv[i], qkv[i]);

    // attention blocks (fused +X residual)
    attn_kernel<<<B_, 256, 0, stream>>>(qkv[0], qkv[1], qkv[2], X, y1, 1);
    attn_kernel<<<B_, 256, 0, stream>>>(qkv[3], qkv[4], qkv[5], X, y2, 5);

    // bn11/bn21 -> stacked activation matrix [128,6000]
    bn_kernel<<<C_, 256, 0, stream>>>(y1, nullptr, bn11g, bn11b, Astk, CL_, 0);
    bn_kernel<<<C_, 256, 0, stream>>>(y2, nullptr, bn21g, bn21b, Astk + 64 * CL_, CL_, 0);

    // linear chain, both blocks batched (weights streamed once each)
    {
        dim3 grid((CL_ + 63) / 64, 1);
        gemm_wmma_kernel<128, 8, 1, 4><<<grid, 256, 0, stream>>>(
            Astk, lin1w, lin1b, Tbuf, 128, CL_, CL_);
        gemm_wmma_kernel<128, 8, 1, 4><<<grid, 256, 0, stream>>>(
            Tbuf, lin2w, lin2b, Obuf, 128, CL_, CL_);
    }

    // bn12/bn22 with fused residual add, written directly into concat layout
    bn_kernel<<<C_, 256, 0, stream>>>(Astk, Obuf, bn12g, bn12b, cat, 2 * CL_, 0);
    bn_kernel<<<C_, 256, 0, stream>>>(Astk + 64 * CL_, Obuf + 64 * CL_,
                                      bn22g, bn22b, cat, 2 * CL_, CL_);

    // converge: out = cat @ cw^T + cb  (cw 288 MB streamed once)
    {
        dim3 grid((CL_ + 127) / 128, 1);
        gemm_wmma_kernel<64, 4, 2, 4><<<grid, 256, 0, stream>>>(
            cat, cw, cb, outp, 64, CL_, 2 * CL_);
    }
}
